// CausalMultiheadSelfAttention_19963007991924
// MI455X (gfx1250) — matrix operand order repaired and checked by reading, not compiled, not run on hardware
//
#include <hip/hip_runtime.h>
#include <math.h>

// ---------------------------------------------------------------------------
// Causal MHA with RoPE on gfx1250 (MI455X). All matmuls on
// v_wmma_f32_16x16x32_bf16 (fp32 accumulate). Flash-style online softmax.
// GEMM wave tile: 32x64 (2x4 WMMA accumulators) to raise WMMA:VMEM density.
// ---------------------------------------------------------------------------

typedef __attribute__((ext_vector_type(16))) __bf16 v16bf;
typedef __attribute__((ext_vector_type(8)))  float  v8f;

union Frag {
    v16bf v;
    uint4 q[2];
};

constexpr int D_MODEL = 1024;
constexpr int SEQ     = 2048;
constexpr int NB      = 4;
constexpr int NH      = 16;
constexpr int DK      = 64;
constexpr int MTOT    = NB * SEQ;                  // 8192
constexpr float LOG_THETA = 9.210340371976184f;    // ln(10000)

// --------------------------- fp32 -> bf16 cast -----------------------------
__global__ void cast_f32_bf16(const float* __restrict__ in,
                              __bf16* __restrict__ out, int n) {
    int i = blockIdx.x * 256 + threadIdx.x;
    if (i < n) out[i] = (__bf16)in[i];
}

__device__ __forceinline__ v8f wmma_bf16(const Frag& a, const Frag& b, v8f c) {
    return __builtin_amdgcn_wmma_f32_16x16x32_bf16(
        false, a.v, false, b.v, (short)0, c, false, false);
}

// --------------------------- GEMM + epilogues ------------------------------
// y[g, o] = sum_i A[g, i] * Bt[o, i]   (A: MTOT x 1024 bf16, Bt: 1024 x 1024 bf16)
// mode 0: RoPE + 0.125 scale, store bf16 Q   at (b,h,s,d)
// mode 1: RoPE,               store bf16 K   at (b,h,s,d)
// mode 2:                     store bf16 V^T at (b,h,d,s)
// mode 3:                     store fp32 to d_out (g, o)
__global__ __launch_bounds__(256)
void gemm_epilogue_kernel(const __bf16* __restrict__ A,
                          const __bf16* __restrict__ Bt,
                          __bf16* __restrict__ o16,
                          float*  __restrict__ o32,
                          const int* __restrict__ tokpos, int mode) {
    const int lane = threadIdx.x & 31;
    const int wave = threadIdx.x >> 5;
    const int ln   = lane & 15;
    const int hi   = (lane >> 4) & 1;
    // block tile 128(M) x 128(N); wave tile 32(M) x 64(N)
    const int m0   = blockIdx.x * 128 + (wave & 3) * 32;
    const int n0   = blockIdx.y * 128 + (wave >> 2) * 64;

    v8f acc[2][4] = {};

    for (int k0 = 0; k0 < D_MODEL; k0 += 32) {
        Frag a[2], b[4];
#pragma unroll
        for (int mt = 0; mt < 2; ++mt) {
            const __bf16* p = A + (size_t)(m0 + mt * 16 + ln) * D_MODEL + k0 + hi * 8;
            a[mt].q[0] = *(const uint4*)p;          // K = kb .. kb+7
            a[mt].q[1] = *(const uint4*)(p + 16);   // K = kb+16 .. kb+23
        }
#pragma unroll
        for (int nt = 0; nt < 4; ++nt) {
            const __bf16* p = Bt + (size_t)(n0 + nt * 16 + ln) * D_MODEL + k0 + hi * 8;
            b[nt].q[0] = *(const uint4*)p;          // K = kb .. kb+7
            b[nt].q[1] = *(const uint4*)(p + 16);   // K = kb+16 .. kb+23
        }
#pragma unroll
        for (int mt = 0; mt < 2; ++mt)
#pragma unroll
            for (int nt = 0; nt < 4; ++nt)
                acc[mt][nt] = wmma_bf16(a[mt], b[nt], acc[mt][nt]);
    }

    // ---- epilogue: C layout row = r + 8*hi, col = ln ----
#pragma unroll
    for (int mt = 0; mt < 2; ++mt) {
#pragma unroll
        for (int nt = 0; nt < 4; ++nt) {
            const int o = n0 + nt * 16 + ln;
#pragma unroll
            for (int r = 0; r < 8; ++r) {
                const int g  = m0 + mt * 16 + hi * 8 + r;
                const int bb = g >> 11;
                const int s  = g & (SEQ - 1);
                float val = acc[mt][nt][r];
                if (mode <= 1) {
                    const int h = o >> 6, d = o & 63, pi = d >> 1;
                    float partner = __shfl_xor(val, 1);   // even<->odd lane pair
                    float pos = (float)tokpos[s];
                    float ang = pos * __expf(-(float)pi * (LOG_THETA / 32.0f));
                    float sn, cs;
                    __sincosf(ang, &sn, &cs);
                    float rot = ((d & 1) == 0) ? (val * cs - partner * sn)
                                               : (partner * sn + val * cs);
                    if (mode == 0) rot *= 0.125f;         // 1/sqrt(dk)
                    o16[(((size_t)(bb * NH + h)) * SEQ + s) * DK + d] = (__bf16)rot;
                } else if (mode == 2) {
                    const int h = o >> 6, d = o & 63;
                    o16[(((size_t)(bb * NH + h)) * DK + d) * SEQ + s] = (__bf16)val;
                } else {
                    o32[(size_t)g * D_MODEL + o] = val;
                }
            }
        }
    }
}

// --------------------------- flash attention -------------------------------
__global__ __launch_bounds__(128)
void attn_kernel(const __bf16* __restrict__ Qb, const __bf16* __restrict__ Kb,
                 const __bf16* __restrict__ Vt, __bf16* __restrict__ Ob) {
    __shared__ __attribute__((aligned(16))) __bf16 lds[4][32][32];

    const int lane = threadIdx.x & 31;
    const int wave = threadIdx.x >> 5;
    const int ln   = lane & 15;
    const int hi   = (lane >> 4) & 1;
    const int bh   = blockIdx.x;
    const int q0   = blockIdx.y * 128 + wave * 32;
    const int b    = bh >> 4;
    const int h    = bh & (NH - 1);

    const __bf16* Qh = Qb + (size_t)bh * SEQ * DK;
    const __bf16* Kh = Kb + (size_t)bh * SEQ * DK;
    const __bf16* Vh = Vt + (size_t)bh * DK * SEQ;

    // Q fragments (already scaled by 1/sqrt(dk)): 2 row tiles x 2 K-steps
    Frag qf[2][2];
#pragma unroll
    for (int mt = 0; mt < 2; ++mt)
#pragma unroll
        for (int kk = 0; kk < 2; ++kk) {
            const __bf16* p = Qh + (size_t)(q0 + mt * 16 + ln) * DK + kk * 32 + hi * 8;
            qf[mt][kk].q[0] = *(const uint4*)p;
            qf[mt][kk].q[1] = *(const uint4*)(p + 16);
        }

    v8f acc[2][4] = {};
    float mrow[2][8], lrow[2][8];
#pragma unroll
    for (int mt = 0; mt < 2; ++mt)
#pragma unroll
        for (int r = 0; r < 8; ++r) {
            mrow[mt][r] = -__builtin_inff();
            lrow[mt][r] = 0.0f;
        }

    const int ntiles = (q0 >> 5) + 1;   // wave-uniform causal key range
    for (int t = 0; t < ntiles; ++t) {
        const int k0 = t * 32;

        // S = Q K^T for this key tile
        Frag kf[2][2];
#pragma unroll
        for (int nt = 0; nt < 2; ++nt)
#pragma unroll
            for (int kk = 0; kk < 2; ++kk) {
                const __bf16* p = Kh + (size_t)(k0 + nt * 16 + ln) * DK + kk * 32 + hi * 8;
                kf[nt][kk].q[0] = *(const uint4*)p;
                kf[nt][kk].q[1] = *(const uint4*)(p + 16);
            }
        v8f sv[2][2] = {};
#pragma unroll
        for (int mt = 0; mt < 2; ++mt)
#pragma unroll
            for (int nt = 0; nt < 2; ++nt) {
                sv[mt][nt] = wmma_bf16(qf[mt][0], kf[nt][0], sv[mt][nt]);
                sv[mt][nt] = wmma_bf16(qf[mt][1], kf[nt][1], sv[mt][nt]);
            }

        // online softmax update (rows split across lane halves; cols = lanes)
#pragma unroll
        for (int mt = 0; mt < 2; ++mt) {
            const int qbase = q0 + mt * 16 + hi * 8;
#pragma unroll
            for (int r = 0; r < 8; ++r) {
                const int qr = qbase + r;
                float s0 = sv[mt][0][r];
                float s1 = sv[mt][1][r];
                if (k0 + ln > qr)      s0 = -__builtin_inff();
                if (k0 + 16 + ln > qr) s1 = -__builtin_inff();
                float tm = fmaxf(s0, s1);
                tm = fmaxf(tm, __shfl_xor(tm, 1));
                tm = fmaxf(tm, __shfl_xor(tm, 2));
                tm = fmaxf(tm, __shfl_xor(tm, 4));
                tm = fmaxf(tm, __shfl_xor(tm, 8));
                const float mold = mrow[mt][r];
                const float mn   = fmaxf(mold, tm);
                const float al   = __expf(mold - mn);
                mrow[mt][r] = mn;
                const float p0 = __expf(s0 - mn);
                const float p1 = __expf(s1 - mn);
                float rs = p0 + p1;
                rs += __shfl_xor(rs, 1);
                rs += __shfl_xor(rs, 2);
                rs += __shfl_xor(rs, 4);
                rs += __shfl_xor(rs, 8);
                lrow[mt][r] = lrow[mt][r] * al + rs;
#pragma unroll
                for (int nt = 0; nt < 4; ++nt) acc[mt][nt][r] *= al;
                // stage P (bf16) in per-wave LDS tile for A-fragment reload
                lds[wave][mt * 16 + hi * 8 + r][ln]      = (__bf16)p0;
                lds[wave][mt * 16 + hi * 8 + r][16 + ln] = (__bf16)p1;
            }
        }

        asm volatile("s_wait_dscnt 0x0" ::: "memory");

        Frag pf[2];
#pragma unroll
        for (int mt = 0; mt < 2; ++mt) {
            const __bf16* lp = &lds[wave][mt * 16 + ln][hi * 8];
            pf[mt].q[0] = *(const uint4*)lp;
            pf[mt].q[1] = *(const uint4*)(lp + 16);
        }

        asm volatile("s_wait_dscnt 0x0" ::: "memory");

        // O += P V   (V^T rows are contiguous in key index)
#pragma unroll
        for (int nt = 0; nt < 4; ++nt) {
            Frag vf;
            const __bf16* p = Vh + (size_t)(nt * 16 + ln) * SEQ + k0 + hi * 8;
            vf.q[0] = *(const uint4*)p;
            vf.q[1] = *(const uint4*)(p + 16);
#pragma unroll
            for (int mt = 0; mt < 2; ++mt)
                acc[mt][nt] = wmma_bf16(pf[mt], vf, acc[mt][nt]);
        }
    }

    // normalize and store bf16 output at (b, s, h*64 + d)
#pragma unroll
    for (int mt = 0; mt < 2; ++mt)
#pragma unroll
        for (int r = 0; r < 8; ++r) {
            const int s = q0 + mt * 16 + hi * 8 + r;
            const float inv = 1.0f / lrow[mt][r];
            const size_t rowbase = (size_t)(b * SEQ + s) * D_MODEL + h * DK;
#pragma unroll
            for (int nt = 0; nt < 4; ++nt)
                Ob[rowbase + nt * 16 + ln] = (__bf16)(acc[mt][nt][r] * inv);
        }
}

// ------------------------------- launcher ----------------------------------
extern "C" void kernel_launch(void* const* d_in, const int* in_sizes, int n_in,
                              void* d_out, int out_size, void* d_ws, size_t ws_size,
                              hipStream_t stream) {
    const float* x   = (const float*)d_in[0];
    const int*   tok = (const int*)d_in[1];
    const float* Wq  = (const float*)d_in[2];
    const float* Wk  = (const float*)d_in[3];
    const float* Wv  = (const float*)d_in[4];
    const float* Wo  = (const float*)d_in[5];
    float* out = (float*)d_out;

    char* w = (char*)d_ws;
    const size_t MB = 1024 * 1024;
    __bf16* xb  = (__bf16*)(w + 0);        // 16 MB
    __bf16* wqb = (__bf16*)(w + 16 * MB);  //  2 MB
    __bf16* wkb = (__bf16*)(w + 18 * MB);
    __bf16* wvb = (__bf16*)(w + 20 * MB);
    __bf16* wob = (__bf16*)(w + 22 * MB);
    __bf16* Qb  = (__bf16*)(w + 24 * MB);  // 16 MB (b,h,s,d)
    __bf16* Kb  = (__bf16*)(w + 40 * MB);  // 16 MB (b,h,s,d)
    __bf16* Vt  = (__bf16*)(w + 56 * MB);  // 16 MB (b,h,d,s)
    __bf16* Ob  = (__bf16*)(w + 72 * MB);  // 16 MB (b*s, d_model)

    const int NX = MTOT * D_MODEL;
    const int NW = D_MODEL * D_MODEL;
    cast_f32_bf16<<<(NX + 255) / 256, 256, 0, stream>>>(x,  xb,  NX);
    cast_f32_bf16<<<(NW + 255) / 256, 256, 0, stream>>>(Wq, wqb, NW);
    cast_f32_bf16<<<(NW + 255) / 256, 256, 0, stream>>>(Wk, wkb, NW);
    cast_f32_bf16<<<(NW + 255) / 256, 256, 0, stream>>>(Wv, wvb, NW);
    cast_f32_bf16<<<(NW + 255) / 256, 256, 0, stream>>>(Wo, wob, NW);

    dim3 gg(MTOT / 128, D_MODEL / 128);
    gemm_epilogue_kernel<<<gg, 256, 0, stream>>>(xb, wqb, Qb, nullptr, tok, 0);
    gemm_epilogue_kernel<<<gg, 256, 0, stream>>>(xb, wkb, Kb, nullptr, tok, 1);
    gemm_epilogue_kernel<<<gg, 256, 0, stream>>>(xb, wvb, Vt, nullptr, tok, 2);

    attn_kernel<<<dim3(NB * NH, SEQ / 128), 128, 0, stream>>>(Qb, Kb, Vt, Ob);

    gemm_epilogue_kernel<<<gg, 256, 0, stream>>>(Ob, wob, nullptr, out, tok, 3);
}
